// FeatureTransformer_73057393705754
// MI455X (gfx1250) — matrix operand order repaired and checked by reading, not compiled, not run on hardware
//
#include <hip/hip_runtime.h>
#include <hip/hip_bf16.h>
#include <stdint.h>

// NNUE feature transformer: y = x @ W^T + b
//   x: [16384, 22528] f32 (0/1 values), W: [128, 22528] f32, b: [128], y: [16384, 128] f32
// HBM-bound on streaming x (1.48 GB @ 23.3 TB/s ~= 64 us floor). Dense bf16 WMMA GEMM
// reaches that floor; fp32->bf16 conversion is exact for x and ~8-bit accurate for W,
// accumulation stays f32 via v_wmma_f32_16x16x32_bf16.

#define INPUT_DIM  22528
#define OUTPUT_DIM 128
#define BATCH      16384
#define KC         64     // K chunk per LDS stage (two 16x16x32 WMMA k-steps)
#define LDA        72     // padded LDS row stride in u16 (144 B, 16B-aligned, conflict-free)
#define M_TILE     64     // rows of x per workgroup -> 256 workgroups

typedef __attribute__((ext_vector_type(16))) __bf16 v16bf;
typedef __attribute__((ext_vector_type(8)))  __bf16 v8bf;
typedef __attribute__((ext_vector_type(4)))  __bf16 v4bf;
typedef __attribute__((ext_vector_type(8)))  float  v8f;
typedef __attribute__((ext_vector_type(4)))  float  v4f;

// Let the backend pick the native fp32->bf16 convert (packed cvt when available)
static __device__ __forceinline__ v4bf pk4(v4f v) {
    return __builtin_convertvector(v, v4bf);
}

static __device__ __forceinline__ v16bf cat16(v8bf lo, v8bf hi) {
    return __builtin_shufflevector(lo, hi, 0,1,2,3,4,5,6,7,8,9,10,11,12,13,14,15);
}

__global__ __launch_bounds__(256) void nnue_ft_wmma(
    const float* __restrict__ x, const float* __restrict__ W,
    const float* __restrict__ bias, float* __restrict__ out)
{
    __shared__ unsigned short sA[M_TILE * LDA];       // x tile, bf16, row-major [m][k]
    __shared__ unsigned short sB[OUTPUT_DIM * LDA];   // W tile, bf16, row-major [n][k]

    const int tid   = threadIdx.x;
    const int lane  = tid & 31;
    const int wave  = tid >> 5;
    const int wm    = wave & 3;      // which 16-row m-tile
    const int wn    = wave >> 2;     // which 64-col n-half
    const int m     = lane & 15;     // row/col within 16x16 tile
    const int half  = lane >> 4;     // lane-half per ISA fragment layouts
    const int mBase = blockIdx.x * M_TILE;

    // global A staging: 64 rows x 64 f32 per chunk -> 4 v4f per thread
    const int ar = tid >> 2;         // row 0..63
    const int ac = tid & 3;          // float4 slot base
    const float* xrow = x + (size_t)(mBase + ar) * INPUT_DIM;

    v8f acc[4] = {};                 // 4 C tiles per wave (16 x 64)

    v4f Areg[4];
    v4f Breg[8];

    // prefetch chunk 0
#pragma unroll
    for (int i = 0; i < 4; ++i)
        Areg[i] = __builtin_nontemporal_load(
            (const v4f*)(xrow + (size_t)(ac + 4 * i) * 4));
#pragma unroll
    for (int i = 0; i < 8; ++i) {
        int l = tid + 256 * i, row = l >> 4, slot = l & 15;
        Breg[i] = *(const v4f*)(W + (size_t)row * INPUT_DIM + slot * 4);
    }

    for (int kb = 0; kb < INPUT_DIM; kb += KC) {
        __syncthreads();  // previous chunk's compute finished; LDS reusable

        // registers -> LDS with fp32->bf16 conversion (ds_store_b64 per v4bf)
#pragma unroll
        for (int i = 0; i < 4; ++i)
            *(v4bf*)(sA + ar * LDA + (ac + 4 * i) * 4) = pk4(Areg[i]);
#pragma unroll
        for (int i = 0; i < 8; ++i) {
            int l = tid + 256 * i, row = l >> 4, slot = l & 15;
            *(v4bf*)(sB + row * LDA + slot * 4) = pk4(Breg[i]);
        }
        __syncthreads();

        // software pipeline: issue next chunk's global loads before compute
        const int kn = kb + KC;
        if (kn < INPUT_DIM) {
#pragma unroll
            for (int i = 0; i < 4; ++i)
                Areg[i] = __builtin_nontemporal_load(
                    (const v4f*)(xrow + kn + (size_t)(ac + 4 * i) * 4));
#pragma unroll
            for (int i = 0; i < 8; ++i) {
                int l = tid + 256 * i, row = l >> 4, slot = l & 15;
                Breg[i] = *(const v4f*)(W + (size_t)row * INPUT_DIM + kn + slot * 4);
            }
        }

        // compute: 2 k-steps x 4 n-tiles = 8 WMMAs per wave per chunk
#pragma unroll
        for (int kk = 0; kk < KC; kk += 32) {
            // A fragment (16x32 bf16): lane(half,m) holds K = kk + 8*half + {0..7, 16..23}
            const unsigned short* pa = sA + (16 * wm + m) * LDA + kk + 8 * half;
            v16bf afrag = cat16(*(const v8bf*)pa, *(const v8bf*)(pa + 16));
#pragma unroll
            for (int t = 0; t < 4; ++t) {
                // B fragment (32x16 bf16): lane(half,n) holds K = kk + 8*half + {0..7, 16..23}
                const unsigned short* pb = sB + (64 * wn + 16 * t + m) * LDA + kk + 8 * half;
                v16bf bfrag = cat16(*(const v8bf*)pb, *(const v8bf*)(pb + 16));
                acc[t] = __builtin_amdgcn_wmma_f32_16x16x32_bf16(
                    false, afrag, false, bfrag, (short)0, acc[t], false, false);
            }
        }
    }

    // epilogue: C layout M = r + 8*half, N = lane&15; add bias, store
#pragma unroll
    for (int t = 0; t < 4; ++t) {
        const int col = 64 * wn + 16 * t + m;
        const float bv = bias[col];
#pragma unroll
        for (int r = 0; r < 8; ++r) {
            const int row = mBase + 16 * wm + r + 8 * half;
            __builtin_nontemporal_store(acc[t][r] + bv,
                                        out + (size_t)row * OUTPUT_DIM + col);
        }
    }
}

extern "C" void kernel_launch(void* const* d_in, const int* in_sizes, int n_in,
                              void* d_out, int out_size, void* d_ws, size_t ws_size,
                              hipStream_t stream) {
    const float* x = (const float*)d_in[0];   // [16384, 22528]
    const float* W = (const float*)d_in[1];   // [128, 22528]
    const float* b = (const float*)d_in[2];   // [128]
    float* out = (float*)d_out;               // [16384, 128]

    dim3 grid(BATCH / M_TILE);                // 256 workgroups
    dim3 block(256);                          // 8 waves (wave32)
    nnue_ft_wmma<<<grid, block, 0, stream>>>(x, W, b, out);
}
